// RetBlock_75453985457153
// MI455X (gfx1250) — compile-verified, matrix operand order checked by reading, not run on hardware
//
#include <hip/hip_runtime.h>
#include <math.h>

// ---------------------------------------------------------------------------
// RetBlock for MI455X (gfx1250): wave32 + v_wmma_f32_16x16x32_f16 everywhere a
// matmul appears (QKV/out/FFN GEMMs, axial-attention QK^T and P@V).
// GEMM inner loop: unguarded b128 fragment loads (weights pre-transposed to
// f16 N x K), 8 WMMAs per wave per K-step (32x64 output tile per wave).
// ---------------------------------------------------------------------------

typedef __attribute__((ext_vector_type(16))) _Float16 v16h;
typedef __attribute__((ext_vector_type(8)))  float    v8f;

#define DEV __device__ __forceinline__

static constexpr int BB  = 16;
static constexpr int HH  = 56;
static constexpr int WW  = 56;
static constexpr int CC  = 256;
static constexpr int NHD = 8;
static constexpr int KD  = 32;
static constexpr int FF  = 1024;
static constexpr int TOK = BB * HH * WW;          // 50176
static constexpr int SEQ = 56;
static constexpr int SEQP = 64;
static constexpr float SCALE_K = 0.17677669529663687f; // 32^-0.5

DEV int laneid() { return (int)(threadIdx.x & 31); }

DEV v8f wmma16(v16h a, v16h b, v8f c) {
  // v_wmma_f32_16x16x32_f16: D = A(16x32) * B(32x16) + C(16x16 f32)
  return __builtin_amdgcn_wmma_f32_16x16x32_f16(false, a, false, b, (short)0, c,
                                                false, false);
}

// Unguarded fragment load: p already points at this lane's first element
// (halves [0..7] at p[0..7], halves [8..15] at p[16..23]). Two b128 loads.
DEV v16h frag_ld(const _Float16* p) {
  v16h f;
#pragma unroll
  for (int i = 0; i < 8; ++i) f[i] = p[i];
#pragma unroll
  for (int i = 0; i < 8; ++i) f[8 + i] = p[16 + i];
  return f;
}

// A fragment (16x32, MxK), guarded rows (attention use). Lanes 0-15: rows
// M=lane, K offsets k0+[0..7],[16..23]; lanes 16-31: K offsets +8 / +24.
DEV v16h load_a(const _Float16* A, int lda, int row0, int k0, int mmax) {
  int lane = laneid();
  int r = row0 + (lane & 15);
  int kb = k0 + ((lane >> 4) << 3);
  v16h a = {};
  if (r < mmax) {
    const _Float16* p = A + (size_t)r * lda + kb;
#pragma unroll
    for (int i = 0; i < 8; ++i) a[i] = p[i];
#pragma unroll
    for (int i = 0; i < 8; ++i) a[8 + i] = p[16 + i];
  }
  return a;
}

// B fragment (32x16, KxN) from row-major B[k*ldb + n]; rows k >= kmax -> zero.
DEV v16h load_b(const _Float16* B, int ldb, int k0, int col0, int kmax) {
  int lane = laneid();
  int c = col0 + (lane & 15);
  int kb = k0 + ((lane >> 4) << 3);
  v16h b = {};
#pragma unroll
  for (int i = 0; i < 8; ++i) {
    int k = kb + i;
    b[i] = (k < kmax) ? B[(size_t)k * ldb + c] : (_Float16)0.f;
  }
#pragma unroll
  for (int i = 0; i < 8; ++i) {
    int k = kb + 16 + i;
    b[8 + i] = (k < kmax) ? B[(size_t)k * ldb + c] : (_Float16)0.f;
  }
  return b;
}

// B = X^T, X row-major (N x K): B[k,n] = X[n*ldx + k]; cols n >= cmax -> zero.
DEV v16h load_bt(const _Float16* X, int ldx, int k0, int col0, int cmax) {
  int lane = laneid();
  int c = col0 + (lane & 15);
  int kb = k0 + ((lane >> 4) << 3);
  v16h b = {};
  if (c < cmax) {
    const _Float16* p = X + (size_t)c * ldx + kb;
#pragma unroll
    for (int i = 0; i < 8; ++i) b[i] = p[i];
#pragma unroll
    for (int i = 0; i < 8; ++i) b[8 + i] = p[16 + i];
  }
  return b;
}

DEV float gelu_exact(float v) {
  return 0.5f * v * (1.f + erff(v * 0.70710678118654752f));
}

// ---------------------------------------------------------------------------
// GEMM: out = act((A[f16 MxK] @ B + bias) * scale) (+ resid)
// B supplied TRANSPOSED as Bt[f16 NxK] so both operands load as contiguous
// b128 fragments. Block: 256 threads = 8 waves; tile 256(M) x 64(N); wave
// computes 32x64 (2 A-frags x 4 B-frags = 8 WMMA per K-step, B reused 2x).
// Requires M % 256 == 0, K % 32 == 0. ACT: 0 = identity, 1 = exact GELU.
// ---------------------------------------------------------------------------
template <int ACT>
__global__ void __launch_bounds__(256)
gemm_wmma(const _Float16* __restrict__ A, const _Float16* __restrict__ Bt,
          const float* __restrict__ bias, float scale,
          const float* __restrict__ resid,
          _Float16* __restrict__ outh, float* __restrict__ outf,
          int M, int N, int K) {
  int wave = (int)(threadIdx.x >> 5);
  int row0 = blockIdx.x * 256 + wave * 32;
  int col0 = blockIdx.y * 64;
  int lane = laneid();
  int koff = (lane >> 4) << 3;

  const _Float16* ap0 = A + (size_t)(row0 + (lane & 15)) * K + koff;
  const _Float16* ap1 = ap0 + (size_t)16 * K;
  const _Float16* bp[4];
#pragma unroll
  for (int nt = 0; nt < 4; ++nt)
    bp[nt] = Bt + (size_t)(col0 + nt * 16 + (lane & 15)) * K + koff;

  v8f acc[2][4] = {};
  for (int k0 = 0; k0 < K; k0 += 32) {
    v16h a0 = frag_ld(ap0 + k0);
    v16h a1 = frag_ld(ap1 + k0);
#pragma unroll
    for (int nt = 0; nt < 4; ++nt) {
      v16h b = frag_ld(bp[nt] + k0);
      acc[0][nt] = wmma16(a0, b, acc[0][nt]);
      acc[1][nt] = wmma16(a1, b, acc[1][nt]);
    }
  }

#pragma unroll
  for (int mt = 0; mt < 2; ++mt) {
    int rb = row0 + mt * 16 + ((lane >> 4) << 3);
#pragma unroll
    for (int nt = 0; nt < 4; ++nt) {
      int col = col0 + nt * 16 + (lane & 15);
      float bv = bias ? bias[col] : 0.f;
#pragma unroll
      for (int i = 0; i < 8; ++i) {
        int r = rb + i;
        float v = (acc[mt][nt][i] + bv) * scale;
        if (ACT == 1) v = gelu_exact(v);
        size_t idx = (size_t)r * N + col;
        if (resid) v += resid[idx];
        if (outf) outf[idx] = v;
        if (outh) outh[idx] = (_Float16)v;
      }
    }
  }
}

// ---------------------------------------------------------------------------
// Elementwise / LN / dwconv kernels
// ---------------------------------------------------------------------------
// Convert + transpose weights: in = f32 (K x N) row-major, out = f16 (N x K).
__global__ void cvt_transpose_f16(const float* __restrict__ in, _Float16* __restrict__ out,
                                  int K, int N) {
  int idx = blockIdx.x * blockDim.x + threadIdx.x;
  if (idx < K * N) {
    int kk = idx / N, nn = idx - kk * N;
    out[(size_t)nn * K + kk] = (_Float16)in[idx];
  }
}

// x1 = x + dwconv3x3(x) ; one block per token, one thread per channel.
__global__ void __launch_bounds__(256)
dwconv3_add(const float* __restrict__ x, const float* __restrict__ w,
            const float* __restrict__ bias, float* __restrict__ out) {
  int c = (int)threadIdx.x;
  int t = (int)blockIdx.x;
  int wq = t % WW, hq = (t / WW) % HH, b = t / (HH * WW);
  float acc = bias[c];
#pragma unroll
  for (int dh = -1; dh <= 1; ++dh) {
    int hh = hq + dh; if (hh < 0 || hh >= HH) continue;
#pragma unroll
    for (int dw = -1; dw <= 1; ++dw) {
      int ww2 = wq + dw; if (ww2 < 0 || ww2 >= WW) continue;
      acc += x[(((size_t)b * HH + hh) * WW + ww2) * CC + c] *
             w[((dh + 1) * 3 + (dw + 1)) * CC + c];
    }
  }
  size_t idx = (size_t)t * CC + c;
  out[idx] = x[idx] + acc;
}

// lepe = dwconv5x5(v_f16) in f32.
__global__ void __launch_bounds__(256)
dwconv5_f16(const _Float16* __restrict__ v, const float* __restrict__ w,
            const float* __restrict__ bias, float* __restrict__ out) {
  int c = (int)threadIdx.x;
  int t = (int)blockIdx.x;
  int wq = t % WW, hq = (t / WW) % HH, b = t / (HH * WW);
  float acc = bias[c];
#pragma unroll
  for (int dh = -2; dh <= 2; ++dh) {
    int hh = hq + dh; if (hh < 0 || hh >= HH) continue;
#pragma unroll
    for (int dw = -2; dw <= 2; ++dw) {
      int ww2 = wq + dw; if (ww2 < 0 || ww2 >= WW) continue;
      acc += (float)v[(((size_t)b * HH + hh) * WW + ww2) * CC + c] *
             w[((dh + 2) * 5 + (dw + 2)) * CC + c];
    }
  }
  out[(size_t)t * CC + c] = acc;
}

// LayerNorm over C=256, f16 output. One block (256 threads) per token.
__global__ void __launch_bounds__(256)
layernorm_f16(const float* __restrict__ x, const float* __restrict__ g,
              const float* __restrict__ bta, _Float16* __restrict__ out) {
  __shared__ float red[256];
  size_t t = blockIdx.x;
  int c = (int)threadIdx.x;
  float v = x[t * CC + c];
  red[c] = v;
  __syncthreads();
  for (int s = 128; s > 0; s >>= 1) { if (c < s) red[c] += red[c + s]; __syncthreads(); }
  float mean = red[0] * (1.f / CC);
  __syncthreads();
  float d = v - mean;
  red[c] = d * d;
  __syncthreads();
  for (int s = 128; s > 0; s >>= 1) { if (c < s) red[c] += red[c + s]; __syncthreads(); }
  float var = red[0] * (1.f / CC);
  float r = rsqrtf(var + 1e-6f);
  out[t * CC + c] = (_Float16)(d * r * g[c] + bta[c]);
}

// ---------------------------------------------------------------------------
// Axial attention (width axis). One block per (b, h, head); 128 threads = 4 waves.
// Scores: Q(56x32) @ K^T via WMMA (K=32 == head dim, single WMMA per tile),
// + mask_w, softmax via LDS, then O = P @ V via WMMA.
// vw layout: [(b*8+head)*3136 + q_w*56 + h]*32 + d  (contiguous (k_h, d) for
// the height pass B-operand).
// ---------------------------------------------------------------------------
__global__ void __launch_bounds__(128)
attn_width(const _Float16* __restrict__ qb, const _Float16* __restrict__ kbuf,
           const _Float16* __restrict__ vb, const float* __restrict__ maskw,
           _Float16* __restrict__ vw) {
  __shared__ float    S[SEQP * SEQP];
  __shared__ _Float16 P[SEQP * SEQP];
  int head = blockIdx.x & 7;
  int h    = (blockIdx.x >> 3) % HH;
  int b    = blockIdx.x / (NHD * HH);
  size_t base = ((size_t)(b * HH + h) * WW) * CC + head * KD;
  const _Float16* qs = qb + base;
  const _Float16* ks = kbuf + base;
  const _Float16* vs = vb + base;
  const float* mh = maskw + (size_t)head * SEQ * SEQ;

  int wave = (int)(threadIdx.x >> 5);
  int row0 = wave * 16;
  int lane = laneid();
  int rb = row0 + ((lane >> 4) << 3);

  // S = Q @ K^T (row stride over w positions = CC)
  v16h aq = load_a(qs, CC, row0, 0, SEQ);
  v8f sc[4];
#pragma unroll
  for (int nt = 0; nt < 4; ++nt) {
    v16h bk = load_bt(ks, CC, 0, nt * 16, SEQ);
    v8f z = {};
    sc[nt] = wmma16(aq, bk, z);
  }
#pragma unroll
  for (int nt = 0; nt < 4; ++nt) {
    int col = nt * 16 + (lane & 15);
#pragma unroll
    for (int i = 0; i < 8; ++i) {
      int r = rb + i;
      float v = (r < SEQ && col < SEQ) ? (sc[nt][i] + mh[r * SEQ + col]) : -1e30f;
      S[r * SEQP + col] = v;
    }
  }
  __syncthreads();

  // Row softmax (thread t handles row t).
  int t = (int)threadIdx.x;
  if (t < SEQP) {
    if (t < SEQ) {
      float m = -1e30f;
      for (int c = 0; c < SEQ; ++c) m = fmaxf(m, S[t * SEQP + c]);
      float s = 0.f;
      for (int c = 0; c < SEQ; ++c) s += __expf(S[t * SEQP + c] - m);
      float inv = 1.f / s;
      for (int c = 0; c < SEQ; ++c)
        P[t * SEQP + c] = (_Float16)(__expf(S[t * SEQP + c] - m) * inv);
      for (int c = SEQ; c < SEQP; ++c) P[t * SEQP + c] = (_Float16)0.f;
    } else {
      for (int c = 0; c < SEQP; ++c) P[t * SEQP + c] = (_Float16)0.f;
    }
  }
  __syncthreads();

  // O = P(64x64) @ V(64x32)
  v8f o[2] = {};
  for (int kk = 0; kk < SEQP; kk += 32) {
    v16h ap = load_a(P, SEQP, row0, kk, SEQP);
#pragma unroll
    for (int nt = 0; nt < 2; ++nt) {
      v16h bv = load_b(vs, CC, kk, nt * 16, SEQ);
      o[nt] = wmma16(ap, bv, o[nt]);
    }
  }
  _Float16* ob = vw + ((size_t)(b * NHD + head) * (HH * WW) + h) * KD;
#pragma unroll
  for (int nt = 0; nt < 2; ++nt) {
    int col = nt * 16 + (lane & 15);
#pragma unroll
    for (int i = 0; i < 8; ++i) {
      int r = rb + i;
      if (r < SEQ) ob[(size_t)r * (WW * KD) + col] = (_Float16)o[nt][i];
    }
  }
}

// ---------------------------------------------------------------------------
// Axial attention (height axis). One block per (b, w, head). Q/K fragment rows
// are strided by WW*CC; V' comes from vw (contiguous). Epilogue fuses + lepe
// and writes the f16 out-projection input.
// ---------------------------------------------------------------------------
__global__ void __launch_bounds__(128)
attn_height(const _Float16* __restrict__ qb, const _Float16* __restrict__ kbuf,
            const _Float16* __restrict__ vwb, const float* __restrict__ maskh,
            const float* __restrict__ lepe, _Float16* __restrict__ proj_in) {
  __shared__ float    S[SEQP * SEQP];
  __shared__ _Float16 P[SEQP * SEQP];
  int head = blockIdx.x & 7;
  int w    = (blockIdx.x >> 3) % WW;
  int b    = blockIdx.x / (NHD * WW);
  size_t base = ((size_t)b * (HH * WW) + w) * CC + head * KD; // h stride = WW*CC
  const _Float16* qs = qb + base;
  const _Float16* ks = kbuf + base;
  const _Float16* vs = vwb + ((size_t)(b * NHD + head) * (HH * WW) + (size_t)w * HH) * KD;
  const float* mh = maskh + (size_t)head * SEQ * SEQ;

  int wave = (int)(threadIdx.x >> 5);
  int row0 = wave * 16;
  int lane = laneid();
  int rb = row0 + ((lane >> 4) << 3);

  v16h aq = load_a(qs, WW * CC, row0, 0, SEQ);
  v8f sc[4];
#pragma unroll
  for (int nt = 0; nt < 4; ++nt) {
    v16h bk = load_bt(ks, WW * CC, 0, nt * 16, SEQ);
    v8f z = {};
    sc[nt] = wmma16(aq, bk, z);
  }
#pragma unroll
  for (int nt = 0; nt < 4; ++nt) {
    int col = nt * 16 + (lane & 15);
#pragma unroll
    for (int i = 0; i < 8; ++i) {
      int r = rb + i;
      float v = (r < SEQ && col < SEQ) ? (sc[nt][i] + mh[r * SEQ + col]) : -1e30f;
      S[r * SEQP + col] = v;
    }
  }
  __syncthreads();

  int t = (int)threadIdx.x;
  if (t < SEQP) {
    if (t < SEQ) {
      float m = -1e30f;
      for (int c = 0; c < SEQ; ++c) m = fmaxf(m, S[t * SEQP + c]);
      float s = 0.f;
      for (int c = 0; c < SEQ; ++c) s += __expf(S[t * SEQP + c] - m);
      float inv = 1.f / s;
      for (int c = 0; c < SEQ; ++c)
        P[t * SEQP + c] = (_Float16)(__expf(S[t * SEQP + c] - m) * inv);
      for (int c = SEQ; c < SEQP; ++c) P[t * SEQP + c] = (_Float16)0.f;
    } else {
      for (int c = 0; c < SEQP; ++c) P[t * SEQP + c] = (_Float16)0.f;
    }
  }
  __syncthreads();

  v8f o[2] = {};
  for (int kk = 0; kk < SEQP; kk += 32) {
    v16h ap = load_a(P, SEQP, row0, kk, SEQP);
#pragma unroll
    for (int nt = 0; nt < 2; ++nt) {
      v16h bv = load_b(vs, KD, kk, nt * 16, SEQ); // V'[k_h, d], ldb = 32
      o[nt] = wmma16(ap, bv, o[nt]);
    }
  }
  // out[b, r(=q_h), w, head*32+d] += lepe ; write f16 projection input.
#pragma unroll
  for (int nt = 0; nt < 2; ++nt) {
    int col = nt * 16 + (lane & 15);
#pragma unroll
    for (int i = 0; i < 8; ++i) {
      int r = rb + i;
      if (r < SEQ) {
        size_t idx = (((size_t)b * HH + r) * WW + w) * CC + head * KD + col;
        proj_in[idx] = (_Float16)(o[nt][i] + lepe[idx]);
      }
    }
  }
}

// ---------------------------------------------------------------------------
// Host-side launch
// ---------------------------------------------------------------------------
extern "C" void kernel_launch(void* const* d_in, const int* in_sizes, int n_in,
                              void* d_out, int out_size, void* d_ws, size_t ws_size,
                              hipStream_t stream) {
  const float* x      = (const float*)d_in[0];
  const float* mask_h = (const float*)d_in[1];
  const float* mask_w = (const float*)d_in[2];
  const float* pos_w  = (const float*)d_in[3];
  const float* pos_b  = (const float*)d_in[4];
  const float* ln1_g  = (const float*)d_in[5];
  const float* ln1_b  = (const float*)d_in[6];
  const float* q_w    = (const float*)d_in[7];
  const float* q_b    = (const float*)d_in[8];
  const float* k_w    = (const float*)d_in[9];
  const float* k_b    = (const float*)d_in[10];
  const float* v_w    = (const float*)d_in[11];
  const float* v_b    = (const float*)d_in[12];
  const float* lepe_w = (const float*)d_in[13];
  const float* lepe_b = (const float*)d_in[14];
  const float* out_w  = (const float*)d_in[15];
  const float* out_b  = (const float*)d_in[16];
  const float* ln2_g  = (const float*)d_in[17];
  const float* ln2_b  = (const float*)d_in[18];
  const float* fc1_w  = (const float*)d_in[19];
  const float* fc1_b  = (const float*)d_in[20];
  const float* fc2_w  = (const float*)d_in[21];
  const float* fc2_b  = (const float*)d_in[22];
  float* out = (float*)d_out;

  const size_t tokC = (size_t)TOK * CC; // 12,845,056 elements
  char* ws = (char*)d_ws;
  size_t off = 0;
  float*    x1      = (float*)(ws + off); off += tokC * 4;
  float*    x2      = (float*)(ws + off); off += tokC * 4;
  _Float16* lnb     = (_Float16*)(ws + off); off += tokC * 2;  // ln1 then ln2
  _Float16* proj_in = (_Float16*)(ws + off); off += tokC * 2;
  _Float16* vwbuf   = (_Float16*)(ws + off); off += tokC * 2;
  _Float16* qf      = (_Float16*)(ws + off); off += tokC * 2;
  _Float16* kf      = (_Float16*)(ws + off); off += tokC * 2;
  _Float16* vf      = (_Float16*)(ws + off); off += tokC * 2;
  float*    lepe    = (float*)(ws + off); off += tokC * 4;
  _Float16* wts     = (_Float16*)(ws + off);
  _Float16* wq   = wts;            // 256x256 (transposed NxK)
  _Float16* wk   = wts + 65536;
  _Float16* wv   = wts + 131072;
  _Float16* wo   = wts + 196608;
  _Float16* wfc1 = wts + 262144;   // 1024x256 (transposed)
  _Float16* wfc2 = wts + 524288;   // 256x1024 (transposed)
  // FFN hidden (50176x1024 f16 = 102.8 MB) aliases dead q/k/v/lepe (128.5 MB).
  _Float16* y1 = qf;

  // 1) weights -> f16, transposed to N x K for contiguous B-fragment loads
  cvt_transpose_f16<<<(65536 + 255) / 256, 256, 0, stream>>>(q_w,   wq,   CC, CC);
  cvt_transpose_f16<<<(65536 + 255) / 256, 256, 0, stream>>>(k_w,   wk,   CC, CC);
  cvt_transpose_f16<<<(65536 + 255) / 256, 256, 0, stream>>>(v_w,   wv,   CC, CC);
  cvt_transpose_f16<<<(65536 + 255) / 256, 256, 0, stream>>>(out_w, wo,   CC, CC);
  cvt_transpose_f16<<<(262144 + 255) / 256, 256, 0, stream>>>(fc1_w, wfc1, CC, FF);
  cvt_transpose_f16<<<(262144 + 255) / 256, 256, 0, stream>>>(fc2_w, wfc2, FF, CC);

  // 2) x1 = x + dwconv3x3(x)
  dwconv3_add<<<TOK, 256, 0, stream>>>(x, pos_w, pos_b, x1);

  // 3) ln1 -> f16
  layernorm_f16<<<TOK, 256, 0, stream>>>(x1, ln1_g, ln1_b, lnb);

  // 4) q / k / v projections (k fused with * KD^-0.5)
  dim3 g256(TOK / 256, CC / 64);
  gemm_wmma<0><<<g256, 256, 0, stream>>>(lnb, wq, q_b, 1.f,      nullptr, qf, nullptr, TOK, CC, CC);
  gemm_wmma<0><<<g256, 256, 0, stream>>>(lnb, wk, k_b, SCALE_K,  nullptr, kf, nullptr, TOK, CC, CC);
  gemm_wmma<0><<<g256, 256, 0, stream>>>(lnb, wv, v_b, 1.f,      nullptr, vf, nullptr, TOK, CC, CC);

  // 5) lepe = dwconv5x5(v)
  dwconv5_f16<<<TOK, 256, 0, stream>>>(vf, lepe_w, lepe_b, lepe);

  // 6) width-axis retention, 7) height-axis retention (+lepe fused)
  attn_width <<<BB * HH * NHD, 128, 0, stream>>>(qf, kf, vf, mask_w, vwbuf);
  attn_height<<<BB * WW * NHD, 128, 0, stream>>>(qf, kf, vwbuf, mask_h, lepe, proj_in);

  // 8) out projection + residual -> x2 (f32)
  gemm_wmma<0><<<g256, 256, 0, stream>>>(proj_in, wo, out_b, 1.f, x1, nullptr, x2, TOK, CC, CC);

  // 9) ln2 -> f16 (reuse ln buffer)
  layernorm_f16<<<TOK, 256, 0, stream>>>(x2, ln2_g, ln2_b, lnb);

  // 10) FFN: fc1 + GELU -> y1 (f16), fc2 + residual -> d_out (f32)
  dim3 gfc1(TOK / 256, FF / 64);
  gemm_wmma<1><<<gfc1, 256, 0, stream>>>(lnb, wfc1, fc1_b, 1.f, nullptr, y1, nullptr, TOK, FF, CC);
  gemm_wmma<0><<<g256, 256, 0, stream>>>(y1, wfc2, fc2_b, 1.f, x2, nullptr, out, TOK, CC, FF);
}
